// mLSTMLayer_42949673567
// MI455X (gfx1250) — matrix operand order repaired and checked by reading, not compiled, not run on hardware
//
#include <hip/hip_runtime.h>
#include <hip/hip_bf16.h>

#define B_    2
#define S_    2048
#define E_    1024
#define H_    1344
#define H2_   2688
#define NH_   4
#define DH_   336
#define DHP_  352     // DH padded to multiple of 32 for WMMA K
#define NPH_  336
#define EPS_  1e-6f
#define KC_   64      // GEMM K-chunk staged in LDS (divides 1024 and 1344)

typedef __attribute__((ext_vector_type(16))) __bf16 v16bf;
typedef __attribute__((ext_vector_type(8)))  __bf16 v8bf;
typedef __attribute__((ext_vector_type(8)))  float  v8f;
typedef __attribute__((ext_vector_type(4)))  unsigned v4u;
typedef __attribute__((ext_vector_type(8)))  unsigned v8u;

// ---------------------------------------------------------------------------
// WMMA fragment loaders (CDNA5 16x16x32 bf16 layouts, ISA 7.12.2)
// ---------------------------------------------------------------------------
static __device__ __forceinline__ v16bf frag_cat(v8bf lo, v8bf hi) {
  return __builtin_shufflevector(lo, hi, 0,1,2,3,4,5,6,7,8,9,10,11,12,13,14,15);
}

static __device__ __forceinline__ v16bf load_a_frag(const __bf16* __restrict__ A, int lda) {
  int lane = threadIdx.x & 31;
  int row  = lane & 15, half = lane >> 4;
  const __bf16* p = A + (size_t)row * lda + half * 8;
  v8bf lo = *(const v8bf*)(p);
  v8bf hi = *(const v8bf*)(p + 16);
  return frag_cat(lo, hi);
}

static __device__ __forceinline__ v16bf load_b_frag(const __bf16* __restrict__ Bt, int ldb) {
  int lane = threadIdx.x & 31;
  int n    = lane & 15, half = lane >> 4;
  const __bf16* p = Bt + (size_t)n * ldb + half * 8;
  v8bf lo = *(const v8bf*)(p);
  v8bf hi = *(const v8bf*)(p + 16);
  return frag_cat(lo, hi);
}

static __device__ __forceinline__ v8f wmma_bf16(v16bf a, v16bf b, v8f c) {
  return __builtin_amdgcn_wmma_f32_16x16x32_bf16(false, a, false, b, (short)0, c, false, false);
}

// Async 16-byte-per-lane copy global -> LDS (ASYNCcnt tracked).
static __device__ __forceinline__ void async_copy_b128(unsigned lds_off, const __bf16* g) {
  asm volatile("global_load_async_to_lds_b128 %0, %1, off"
               :: "v"(lds_off), "v"(g) : "memory");
}
static __device__ __forceinline__ void wait_async0() {
  asm volatile("s_wait_asynccnt 0x0" ::: "memory");
}

// ---------------------------------------------------------------------------
// Tensor Data Mover: 2D tile (elem = 2 bytes) global -> LDS.
// D# per CDNA5 ISA:
//   group0: count=1 | lds_addr | global_addr(57b) | type=2
//   group1: data_size=1(2B) | tensor_dim0/1 | tile_dim0/1 | tensor_dim0_stride
// Issued wave-uniform (SGPR groups); tracked on TENSORcnt.
// ---------------------------------------------------------------------------
static __device__ __forceinline__ void tdm_load_2d_b16(
    unsigned lds_byte, unsigned long long gaddr,
    unsigned tile_d0, unsigned tile_d1,
    unsigned tensor_d0, unsigned tensor_d1,
    unsigned long long stride0_elems)
{
  v4u g0;
  g0[0] = 1u;                                   // count=1, user mode
  g0[1] = lds_byte;                             // LDS dest (bytes)
  g0[2] = (unsigned)gaddr;                      // global_addr[31:0]
  g0[3] = (unsigned)((gaddr >> 32) & 0x1FFFFFFull) | (2u << 30);  // addr[56:32] | type=2
  v8u g1;
  g1[0] = 1u << 16;                             // data_size = 2 bytes
  g1[1] = (tensor_d0 & 0xFFFFu) << 16;          // abar_addr=0 | tensor_dim0[15:0]
  g1[2] = (tensor_d0 >> 16) | ((tensor_d1 & 0xFFFFu) << 16);
  g1[3] = (tensor_d1 >> 16) | (tile_d0 << 16);  // tensor_dim1[31:16] | tile_dim0
  g1[4] = tile_d1;                              // tile_dim1 | tile_dim2=0
  g1[5] = (unsigned)stride0_elems;              // tensor_dim0_stride[31:0]
  g1[6] = (unsigned)((stride0_elems >> 32) & 0xFFFFull);  // stride[47:32] | dim1_stride lo = 0
  g1[7] = 0u;
  asm volatile("tensor_load_to_lds %0, %1" :: "s"(g0), "s"(g1) : "memory");
}

// ---------------------------------------------------------------------------
// Pipelined bf16 GEMM: C[M,N] f32 = A[M,K] * Bt[N,K] (B pre-transposed).
// Block = 4 waves -> one 16x64 tile. K consumed in KC=64 chunks,
// double-buffered in LDS: A tile via TDM tensor_load_to_lds (wave 0),
// B tile via per-lane global_load_async_to_lds_b128. The DMA engines fetch
// chunk c+1 while WMMAs consume chunk c from LDS (ds_load_b128).
// ---------------------------------------------------------------------------
__global__ void gemm_bf16_async_kernel(const __bf16* __restrict__ A,
                                       const __bf16* __restrict__ Bt,
                                       float* __restrict__ C,
                                       int M, int N, int Kdim) {
  __shared__ __bf16 sA[2][16 * KC_];
  __shared__ __bf16 sB[2][64 * KC_];

  int tid  = threadIdx.x;
  int wave = tid >> 5;
  int ngroups = N >> 6;
  int mt = blockIdx.x / ngroups;
  int ng = blockIdx.x % ngroups;
  if (mt * 16 >= M) return;

  const __bf16* Abase = A  + (size_t)mt * 16 * Kdim;
  const __bf16* Bbase = Bt + (size_t)ng * 64 * Kdim;
  int nch = Kdim / KC_;

  // Hoisted per-thread issue addressing: thread owns 4 B slots
  //   row = tid/8 + 16*i, col = (tid&7)*8, lds byte = tid*16 + i*2048.
  const __bf16* gB = Bbase + (size_t)(tid >> 3) * Kdim + (tid & 7) * 8;
  unsigned ldsA[2] = { (unsigned)(size_t)&sA[0][0], (unsigned)(size_t)&sA[1][0] };
  unsigned ldsB[2] = { (unsigned)(size_t)&sB[0][0] + (unsigned)tid * 16u,
                       (unsigned)(size_t)&sB[1][0] + (unsigned)tid * 16u };

  auto issue_chunk = [&](int k0, int buf) {
    if (wave == 0)
      tdm_load_2d_b16(ldsA[buf], (unsigned long long)(size_t)(Abase + k0),
                      (unsigned)KC_, 16u, (unsigned)Kdim, 16u,
                      (unsigned long long)Kdim);
#pragma unroll
    for (int i = 0; i < 4; ++i)
      async_copy_b128(ldsB[buf] + (unsigned)i * 2048u, gB + (size_t)i * 16 * Kdim + k0);
  };

  issue_chunk(0, 0);
  if (wave == 0) __builtin_amdgcn_s_wait_tensorcnt((short)0);
  wait_async0();
  __syncthreads();

  v8f acc = (v8f){0,0,0,0,0,0,0,0};
  for (int c = 0; c < nch; ++c) {
    int cur = c & 1;
    if (c + 1 < nch) issue_chunk((c + 1) * KC_, 1 - cur);
#pragma unroll
    for (int k0 = 0; k0 < KC_; k0 += 32) {
      v16bf a = load_a_frag(&sA[cur][k0], KC_);
      v16bf b = load_b_frag(&sB[cur][(wave * 16) * KC_ + k0], KC_);
      acc = wmma_bf16(a, b, acc);
    }
    if (wave == 0) __builtin_amdgcn_s_wait_tensorcnt((short)0);
    wait_async0();       // chunk c+1 landed in LDS
    __syncthreads();     // everyone done reading cur; next buffer published
  }

  int lane = tid & 31;
  int n = lane & 15, half = lane >> 4;
  float* Cp = C + (size_t)(mt * 16) * N + ng * 64 + wave * 16 + n;
#pragma unroll
  for (int j = 0; j < 8; ++j)
    Cp[(size_t)(half * 8 + j) * N] = acc[j];
}

// ---------------------------------------------------------------------------
// f32 -> bf16 conversion and transposed conversion
// ---------------------------------------------------------------------------
__global__ void cvt_bf16_kernel(const float* __restrict__ src, __bf16* __restrict__ dst, int n) {
  int i = blockIdx.x * blockDim.x + threadIdx.x;
  if (i < n) dst[i] = (__bf16)src[i];
}

__global__ void transpose_bf16_kernel(const float* __restrict__ src, __bf16* __restrict__ dst,
                                      int rows, int cols) {
  int i = blockIdx.x * blockDim.x + threadIdx.x;
  if (i >= rows * cols) return;
  int r = i / cols, c = i % cols;
  dst[(size_t)c * rows + r] = (__bf16)src[i];
}

// ---------------------------------------------------------------------------
// Fused: causal depthwise conv(K=4) + SiLU + headwise q/k/v (PD=4) +
// gate GEMVs (3H x 4) + skip*act + silu(z). One block per (b,s) row.
// ---------------------------------------------------------------------------
__global__ void fused_conv_qkv_gates_kernel(
    const float* __restrict__ Xin,   // (B*S, 2H)
    const float* __restrict__ convk, // (4, H)
    const float* __restrict__ convb, // (H)
    const float* __restrict__ Wq, const float* __restrict__ Wk, const float* __restrict__ Wv, // (NPH,4,4)
    const float* __restrict__ Wig, const float* __restrict__ big,   // (3H,4),(4)
    const float* __restrict__ Wfg, const float* __restrict__ bfg,
    const float* __restrict__ skip,
    __bf16* __restrict__ qbf, __bf16* __restrict__ kbf,  // (B,NH,S,DHP)
    __bf16* __restrict__ vT,                             // (B,NH,DH,S)
    float* __restrict__ sact, float* __restrict__ sz,    // (B*S, H)
    float* __restrict__ igv, float* __restrict__ fgv)    // (B,NH,S)
{
  __shared__ float s_act[H_];
  __shared__ float s_xm[H_];
  __shared__ float s_q[H_], s_k[H_], s_v[H_];
  __shared__ float s_part[8][8];   // [wave][gate]

  int row = blockIdx.x;
  int b = row / S_, s = row % S_;
  int tid = threadIdx.x;
  int wave = tid >> 5, lane = tid & 31;

  for (int ch = tid; ch < H_; ch += blockDim.x) {
    float xm = Xin[(size_t)row * H2_ + ch];
    float acc = convb[ch];
#pragma unroll
    for (int j = 0; j < 4; ++j) {
      int sp = s - 3 + j;
      float xv = (sp >= 0) ? Xin[((size_t)(b * S_ + sp)) * H2_ + ch] : 0.f;
      acc += xv * convk[j * H_ + ch];
    }
    float a = acc / (1.f + __expf(-acc));     // silu
    s_act[ch] = a;
    s_xm[ch]  = xm;
    sact[(size_t)row * H_ + ch] = skip[ch] * a;
    float z = Xin[(size_t)row * H2_ + H_ + ch];
    sz[(size_t)row * H_ + ch] = z / (1.f + __expf(-z));
  }
  __syncthreads();

  // headwise block-diagonal projections, PD=4
  for (int h = tid; h < NPH_; h += blockDim.x) {
    float a0 = s_act[4*h], a1 = s_act[4*h+1], a2 = s_act[4*h+2], a3 = s_act[4*h+3];
    float x0 = s_xm[4*h],  x1 = s_xm[4*h+1],  x2 = s_xm[4*h+2],  x3 = s_xm[4*h+3];
    const float* wq = Wq + h * 16;
    const float* wk = Wk + h * 16;
    const float* wv = Wv + h * 16;
#pragma unroll
    for (int o = 0; o < 4; ++o) {
      float qv = a0*wq[o] + a1*wq[4+o] + a2*wq[8+o] + a3*wq[12+o];
      float kv = a0*wk[o] + a1*wk[4+o] + a2*wk[8+o] + a3*wk[12+o];
      float vv = x0*wv[o] + x1*wv[4+o] + x2*wv[8+o] + x3*wv[12+o];
      int c = 4 * h + o;
      s_q[c] = qv; s_k[c] = kv; s_v[c] = vv;
      int nh = c / DH_, d = c % DH_;
      size_t qk_off = ((size_t)(b * NH_ + nh) * S_ + s) * DHP_ + d;
      qbf[qk_off] = (__bf16)qv;
      kbf[qk_off] = (__bf16)kv;
      vT[((size_t)(b * NH_ + nh) * DH_ + d) * S_ + s] = (__bf16)vv;
    }
  }
  // zero q/k pad region d in [DH, DHP)
  for (int i = tid; i < NH_ * (DHP_ - DH_); i += blockDim.x) {
    int nh = i / (DHP_ - DH_);
    int d  = DH_ + i % (DHP_ - DH_);
    size_t off = ((size_t)(b * NH_ + nh) * S_ + s) * DHP_ + d;
    qbf[off] = (__bf16)0.f;
    kbf[off] = (__bf16)0.f;
  }
  __syncthreads();

  // gates: gin = [q,k,v] (3H) dot W_ig/W_fg (3H x 4); deterministic tree reduce
  float pig[4] = {0,0,0,0}, pfg[4] = {0,0,0,0};
  for (int c = tid; c < H_; c += blockDim.x) {
    float qv = s_q[c], kv = s_k[c], vv = s_v[c];
#pragma unroll
    for (int n = 0; n < 4; ++n) {
      pig[n] += qv * Wig[(size_t)c * NH_ + n]
              + kv * Wig[(size_t)(H_ + c) * NH_ + n]
              + vv * Wig[(size_t)(2 * H_ + c) * NH_ + n];
      pfg[n] += qv * Wfg[(size_t)c * NH_ + n]
              + kv * Wfg[(size_t)(H_ + c) * NH_ + n]
              + vv * Wfg[(size_t)(2 * H_ + c) * NH_ + n];
    }
  }
#pragma unroll
  for (int n = 0; n < 4; ++n) {
#pragma unroll
    for (int m = 1; m < 32; m <<= 1) {
      pig[n] += __shfl_xor(pig[n], m, 32);
      pfg[n] += __shfl_xor(pfg[n], m, 32);
    }
  }
  if (lane == 0) {
#pragma unroll
    for (int n = 0; n < 4; ++n) { s_part[wave][n] = pig[n]; s_part[wave][4 + n] = pfg[n]; }
  }
  __syncthreads();
  if (tid < 8) {
    float t = 0.f;
#pragma unroll
    for (int w = 0; w < 8; ++w) t += s_part[w][tid];
    if (tid < 4) igv[((size_t)(b * NH_ + tid)) * S_ + s] = t + big[tid];
    else         fgv[((size_t)(b * NH_ + (tid - 4))) * S_ + s] = t + bfg[tid - 4];
  }
}

// ---------------------------------------------------------------------------
// Sequential gate scan per (b,nh): c = cumsum(logsigmoid(fg)),
// b_t = ig_t - c_t, pm_s = prefix-max of b_t.  (tiny: 8 x 2048)
// ---------------------------------------------------------------------------
__global__ void gate_scan_kernel(const float* __restrict__ igv, const float* __restrict__ fgv,
                                 float* __restrict__ cvec, float* __restrict__ bvec,
                                 float* __restrict__ pmvec) {
  int g = threadIdx.x;
  if (g >= B_ * NH_) return;
  const float* igp = igv + (size_t)g * S_;
  const float* fgp = fgv + (size_t)g * S_;
  float c = 0.f, pm = -1e30f;
  for (int s = 0; s < S_; ++s) {
    float f  = fgp[s];
    float lf = fminf(f, 0.f) - log1pf(__expf(-fabsf(f)));  // stable logsigmoid
    c += lf;
    float bt = igp[s] - c;
    pm = fmaxf(pm, bt);
    cvec[(size_t)g * S_ + s]  = c;
    bvec[(size_t)g * S_ + s]  = bt;
    pmvec[(size_t)g * S_ + s] = pm;
  }
}

// ---------------------------------------------------------------------------
// Streamed mLSTM cell: 1 wave per (b,nh,16-row tile). Per 32-col chunk:
// QK^T (bf16 WMMA, K=DHP) -> decay/mask in f32 -> bf16 C via LDS reshape ->
// C @ V (bf16 WMMA over 21 N-tiles). Exact stabilizer m = c_s + pm_s.
// ---------------------------------------------------------------------------
__global__ void mlstm_attn_kernel(const __bf16* __restrict__ qbf,
                                  const __bf16* __restrict__ kbf,
                                  const __bf16* __restrict__ vT,
                                  const float* __restrict__ cvec,
                                  const float* __restrict__ bvec,
                                  const float* __restrict__ pmvec,
                                  float* __restrict__ hout)   // (B,NH,S,DH)
{
  __shared__ __bf16 lds_c[16 * 32];

  int blk = blockIdx.x;
  int rt  = blk & (S_ / 16 - 1);
  int g   = blk / (S_ / 16);
  int s0  = rt * 16;
  int lane = threadIdx.x & 31;
  int n = lane & 15, half = lane >> 4;

  const __bf16* Q = qbf + (size_t)g * S_ * DHP_;
  const __bf16* K = kbf + (size_t)g * S_ * DHP_;
  const __bf16* V = vT  + (size_t)g * DH_ * S_;
  const float*  bv = bvec  + (size_t)g * S_;
  const float*  pv = pmvec + (size_t)g * S_;
  const float*  cv = cvec  + (size_t)g * S_;

  float pm_r[8], rowsum[8];
#pragma unroll
  for (int j = 0; j < 8; ++j) {
    pm_r[j]   = pv[s0 + half * 8 + j];
    rowsum[j] = 0.f;
  }

  v8f acc[21];
#pragma unroll
  for (int t = 0; t < 21; ++t) acc[t] = (v8f){0,0,0,0,0,0,0,0};

  const float scale = rsqrtf((float)DH_);

  for (int jj = 0; jj <= s0 + 15; jj += 32) {
    // ---- QK^T: two 16x16 tiles over K = DHP
    v8f qk0 = (v8f){0,0,0,0,0,0,0,0};
    v8f qk1 = (v8f){0,0,0,0,0,0,0,0};
    for (int k0 = 0; k0 < DHP_; k0 += 32) {
      v16bf a  = load_a_frag(Q + (size_t)s0 * DHP_ + k0, DHP_);
      v16bf b0 = load_b_frag(K + (size_t)jj * DHP_ + k0, DHP_);
      v16bf b1 = load_b_frag(K + (size_t)(jj + 16) * DHP_ + k0, DHP_);
      qk0 = wmma_bf16(a, b0, qk0);
      qk1 = wmma_bf16(a, b1, qk1);
    }
    // ---- decay + causal mask, running rowsum; stage bf16 C tile in LDS
#pragma unroll
    for (int j = 0; j < 8; ++j) {
      int srow = s0 + half * 8 + j;
      int t0 = jj + n, t1 = jj + 16 + n;
      float d0 = (t0 <= srow) ? __expf(bv[t0] - pm_r[j]) : 0.f;
      float d1 = (t1 <= srow) ? __expf(bv[t1] - pm_r[j]) : 0.f;
      float c0 = scale * qk0[j] * d0;
      float c1 = scale * qk1[j] * d1;
      rowsum[j] += c0 + c1;
      lds_c[(half * 8 + j) * 32 + n]      = (__bf16)c0;
      lds_c[(half * 8 + j) * 32 + 16 + n] = (__bf16)c1;
    }
    __syncthreads();
    // ---- reload C as A-fragment (16x32, lda=32) from LDS
    v16bf ca;
    {
      int arow = lane & 15, ah = lane >> 4;
      const __bf16* p0 = &lds_c[arow * 32 + ah * 8];
      const __bf16* p1 = &lds_c[arow * 32 + 16 + ah * 8];
#pragma unroll
      for (int i = 0; i < 8; ++i) ca[i] = p0[i];
#pragma unroll
      for (int i = 0; i < 8; ++i) ca[8 + i] = p1[i];
    }
    // ---- C @ V over 21 N-tiles of DH (V^T rows are contiguous in t)
#pragma unroll
    for (int nt = 0; nt < 21; ++nt) {
      const __bf16* p = V + (size_t)(nt * 16 + n) * S_ + jj + half * 8;
      v8bf lo = *(const v8bf*)(p);
      v8bf hi = *(const v8bf*)(p + 16);
      v16bf vb = frag_cat(lo, hi);
      acc[nt] = wmma_bf16(ca, vb, acc[nt]);
    }
    __syncthreads();
  }

  // reduce rowsum across the 16 lanes of each half-group
#pragma unroll
  for (int j = 0; j < 8; ++j) {
    float r = rowsum[j];
#pragma unroll
    for (int m = 1; m < 16; m <<= 1) r += __shfl_xor(r, m, 32);
    rowsum[j] = r;
  }

  // normalize and write h
#pragma unroll
  for (int j = 0; j < 8; ++j) {
    int srow = s0 + half * 8 + j;
    float m  = cv[srow] + pm_r[j];
    float nr = fmaxf(fabsf(rowsum[j]), __expf(-m)) + EPS_;
    float inv = 1.f / nr;
#pragma unroll
    for (int nt = 0; nt < 21; ++nt)
      hout[((size_t)g * S_ + srow) * DH_ + nt * 16 + n] = acc[nt][j] * inv;
  }
}

// ---------------------------------------------------------------------------
// Per-head groupnorm (scale only) + skip + silu(z) gating -> bf16 h_state
// ---------------------------------------------------------------------------
__global__ void norm_combine_kernel(const float* __restrict__ hout,  // (B,NH,S,DH)
                                    const float* __restrict__ sact,
                                    const float* __restrict__ sz,
                                    const float* __restrict__ lnw,
                                    __bf16* __restrict__ hstate) {   // (B*S, H)
  int row = blockIdx.x;
  int b = row / S_, s = row % S_;
  int wave = threadIdx.x >> 5, lane = threadIdx.x & 31;
  const float* hp = hout + (((size_t)(b * NH_ + wave)) * S_ + s) * DH_;
  float sum = 0.f, sq = 0.f;
  for (int d = lane; d < DH_; d += 32) { float v = hp[d]; sum += v; sq += v * v; }
#pragma unroll
  for (int m = 1; m < 32; m <<= 1) {
    sum += __shfl_xor(sum, m, 32);
    sq  += __shfl_xor(sq,  m, 32);
  }
  float mu   = sum / (float)DH_;
  float var  = sq / (float)DH_ - mu * mu;
  float rstd = rsqrtf(var + 1e-5f);
  for (int d = lane; d < DH_; d += 32) {
    int ch = wave * DH_ + d;
    float hn = (hp[d] - mu) * rstd * lnw[ch];
    float hs = (hn + sact[(size_t)row * H_ + ch]) * sz[(size_t)row * H_ + ch];
    hstate[(size_t)row * H_ + ch] = (__bf16)hs;
  }
}

// ---------------------------------------------------------------------------
extern "C" void kernel_launch(void* const* d_in, const int* in_sizes, int n_in,
                              void* d_out, int out_size, void* d_ws, size_t ws_size,
                              hipStream_t stream) {
  const float* x      = (const float*)d_in[0];
  const float* W_up   = (const float*)d_in[1];
  const float* conv_k = (const float*)d_in[2];
  const float* conv_b = (const float*)d_in[3];
  const float* Wq     = (const float*)d_in[4];
  const float* Wk     = (const float*)d_in[5];
  const float* Wv     = (const float*)d_in[6];
  const float* Wig    = (const float*)d_in[7];
  const float* big    = (const float*)d_in[8];
  const float* Wfg    = (const float*)d_in[9];
  const float* bfg    = (const float*)d_in[10];
  const float* lnw    = (const float*)d_in[11];
  const float* skip   = (const float*)d_in[12];
  const float* W_down = (const float*)d_in[13];

  char* ws = (char*)d_ws;
  auto alloc = [&](size_t bytes) -> char* {
    char* p = ws;
    ws += (bytes + 255) & ~(size_t)255;
    return p;
  };
  const int BS = B_ * S_;

  __bf16* xbf    = (__bf16*)alloc((size_t)BS * E_ * 2);
  __bf16* Wupt   = (__bf16*)alloc((size_t)H2_ * E_ * 2);
  __bf16* Wdt    = (__bf16*)alloc((size_t)E_ * H_ * 2);
  float*  Xinner = (float*) alloc((size_t)BS * H2_ * 4);
  __bf16* qbf    = (__bf16*)alloc((size_t)B_ * NH_ * S_ * DHP_ * 2);
  __bf16* kbf    = (__bf16*)alloc((size_t)B_ * NH_ * S_ * DHP_ * 2);
  __bf16* vT     = (__bf16*)alloc((size_t)B_ * NH_ * DH_ * S_ * 2);
  float*  sact   = (float*) alloc((size_t)BS * H_ * 4);
  float*  sz     = (float*) alloc((size_t)BS * H_ * 4);
  float*  igv    = (float*) alloc((size_t)B_ * NH_ * S_ * 4);
  float*  fgv    = (float*) alloc((size_t)B_ * NH_ * S_ * 4);
  float*  cvec   = (float*) alloc((size_t)B_ * NH_ * S_ * 4);
  float*  bvec   = (float*) alloc((size_t)B_ * NH_ * S_ * 4);
  float*  pmvec  = (float*) alloc((size_t)B_ * NH_ * S_ * 4);
  float*  hbuf   = (float*) alloc((size_t)B_ * NH_ * S_ * DH_ * 4);
  __bf16* hstate = (__bf16*)alloc((size_t)BS * H_ * 2);

  // 1) conversions / transposes
  {
    int n = BS * E_;
    cvt_bf16_kernel<<<(n + 255) / 256, 256, 0, stream>>>(x, xbf, n);
  }
  {
    int n = E_ * H2_;
    transpose_bf16_kernel<<<(n + 255) / 256, 256, 0, stream>>>(W_up, Wupt, E_, H2_);
  }
  {
    int n = H_ * E_;
    transpose_bf16_kernel<<<(n + 255) / 256, 256, 0, stream>>>(W_down, Wdt, H_, E_);
  }

  // 2) up-projection GEMM: (BS x 2H) = (BS x E) * (E x 2H), TDM+async pipelined
  {
    int blocks = (BS / 16) * (H2_ / 64);           // 256 * 42
    gemm_bf16_async_kernel<<<blocks, 128, 0, stream>>>(xbf, Wupt, Xinner, BS, H2_, E_);
  }

  // 3) fused conv + silu + qkv + gates
  fused_conv_qkv_gates_kernel<<<BS, 256, 0, stream>>>(
      Xinner, conv_k, conv_b, Wq, Wk, Wv, Wig, big, Wfg, bfg, skip,
      qbf, kbf, vT, sact, sz, igv, fgv);

  // 4) gate scan
  gate_scan_kernel<<<1, 32, 0, stream>>>(igv, fgv, cvec, bvec, pmvec);

  // 5) streamed mLSTM attention: 1 wave per (b,nh,row-tile)
  mlstm_attn_kernel<<<B_ * NH_ * (S_ / 16), 32, 0, stream>>>(
      qbf, kbf, vT, cvec, bvec, pmvec, hbuf);

  // 6) groupnorm + skip + gating
  norm_combine_kernel<<<BS, 128, 0, stream>>>(hbuf, sact, sz, lnw, hstate);

  // 7) down-projection GEMM: (BS x E) = (BS x H) * (H x E), TDM+async pipelined
  {
    int blocks = (BS / 16) * (E_ / 64);            // 256 * 16
    gemm_bf16_async_kernel<<<blocks, 128, 0, stream>>>(hstate, Wdt, (float*)d_out, BS, E_, H_);
  }
}
